// ImageCaptioningModel_8916352106712
// MI455X (gfx1250) — matrix operand order repaired and checked by reading, not compiled, not run on hardware
//
#include <hip/hip_runtime.h>
#include <hip/hip_bf16.h>
#include <math.h>

// ---------------------------------------------------------------------------
// Problem constants (from reference): B,N,D,H,K,L = 32,36,512,4,5,3
// ---------------------------------------------------------------------------
namespace {
constexpr int Bz  = 32;
constexpr int Nn_ = 36;
constexpr int Dd  = 512;
constexpr int Hh  = 4;
constexpr int Kk  = 5;
constexpr int Ll  = 3;
constexpr int Cc  = Dd / Hh;      // 128
constexpr int EMBd = 256;
constexpr int VOC  = 30000;
constexpr int Tt   = 20;
constexpr int SDd  = 7;
constexpr int Ee   = Nn_ * Kk;    // 180
}

typedef __attribute__((ext_vector_type(16))) _Float16 v16h;
typedef __attribute__((ext_vector_type(8)))  float    v8f;

// ---------------------------------------------------------------------------
// f32 -> f16 conversion (weights converted once per launch)
// ---------------------------------------------------------------------------
__global__ void k_f32_to_f16(const float* __restrict__ src,
                             _Float16* __restrict__ dst, int n)
{
  int i = blockIdx.x * blockDim.x + threadIdx.x;
  if (i < n) dst[i] = (_Float16)src[i];
}

// ---------------------------------------------------------------------------
// WMMA GEMM:  Y[M x Nc] = A[M x Kd](f32) @ Wh[Nc x Kd](f16)^T (+bias) (+=Y)
// One wave handles one 16-wide N tile and TWO 16-row M tiles (static, no
// guards): the B fragment is loaded once per k-chunk and reused for 2 WMMAs.
// Requires: M % 32 == 0, Nc % 16 == 0, Kd % 32 == 0 (true for all call sites:
//   M in {1152, 5760, 32, 640}).
// Optional rowIdx: A row m is A[rowIdx[m*ridxStride]] (embedding gather).
// Fragment layouts used here (wave32; A and B share contiguous k order):
//   A 16x32 f16 : lane half h, m=lane&15; slot 2v holds K = 16*h + 2*v
//   B 32x16 f16 : lane half h, n=lane&15; slot 2v holds K = 16*h + 2*v
//                 -> contiguous run [16*h, 16*h+16): single 32-byte vector load
//   C/D 16x16 f32: lane half h, n=lane&15; slot v holds M = v + 8*h
// Register budget: 16 acc + 16 A-frag + 8 B-frag + pointers ~= 56 VGPRs
// (spill-free; round-2 MBLK=4 + dynamic guard variant spilled to scratch).
// ---------------------------------------------------------------------------
__global__ void k_wmma_gemm(const float* __restrict__ A, int lda,
                            const int* __restrict__ rowIdx, int ridxStride,
                            const _Float16* __restrict__ Wh, int ldw,
                            const float* __restrict__ bias,
                            float* __restrict__ Y, int ldy,
                            int M, int Nc, int Kd, int accum)
{
  const int lane  = threadIdx.x & 31;
  const int wid   = threadIdx.x >> 5;
  const int tileN = blockIdx.x * 4 + wid;
  if (tileN * 16 >= Nc) return;          // wave-uniform exit
  const int half = lane >> 4;
  const int l16  = lane & 15;

  const int m0 = blockIdx.y * 32 + l16;  // first row tile
  const int m1 = m0 + 16;                // second row tile
  const int arow0 = rowIdx ? rowIdx[(size_t)m0 * ridxStride] : m0;
  const int arow1 = rowIdx ? rowIdx[(size_t)m1 * ridxStride] : m1;
  const float* __restrict__ Ar0 = A + (size_t)arow0 * lda;
  const float* __restrict__ Ar1 = A + (size_t)arow1 * lda;
  const _Float16* __restrict__ Wr =
      Wh + (size_t)(tileN * 16 + l16) * ldw + 16 * half;

  v8f acc0 = {0.f, 0.f, 0.f, 0.f, 0.f, 0.f, 0.f, 0.f};
  v8f acc1 = acc0;

  for (int k0 = 0; k0 < Kd; k0 += 32) {
    // whole B fragment in one 32B vector load (2x global_load_b128, no cvt)
    v16h b = *(const v16h*)(Wr + k0);
    if (k0 + 32 < Kd) __builtin_prefetch(Wr + k0 + 32, 0, 1); // global_prefetch
    v16h a0, a1;
#pragma unroll
    for (int v = 0; v < 8; ++v) {
      const int ka = 16 * half + 2 * v;
      a0[2 * v]     = (_Float16)Ar0[k0 + ka];
      a0[2 * v + 1] = (_Float16)Ar0[k0 + ka + 1];
      a1[2 * v]     = (_Float16)Ar1[k0 + ka];
      a1[2 * v + 1] = (_Float16)Ar1[k0 + ka + 1];
    }
    acc0 = __builtin_amdgcn_wmma_f32_16x16x32_f16(
        false, a0, false, b, (short)0, acc0, false, false);
    acc1 = __builtin_amdgcn_wmma_f32_16x16x32_f16(
        false, a1, false, b, (short)0, acc1, false, false);
  }

  const int n = tileN * 16 + l16;
  const float bval = bias ? bias[n] : 0.f;
#pragma unroll
  for (int v = 0; v < 8; ++v) {
    const int mA = blockIdx.y * 32 + v + 8 * half;
    const int mB = mA + 16;
    const size_t oA = (size_t)mA * ldy + n;
    const size_t oB = (size_t)mB * ldy + n;
    float vA = acc0[v] + bval;
    float vB = acc1[v] + bval;
    if (accum) { vA += Y[oA]; vB += Y[oB]; }
    Y[oA] = vA;
    Y[oB] = vB;
  }
}

// ---------------------------------------------------------------------------
// Pairwise spatial features sf[b,i,j,7] and distance d2[b,i,j]
// ---------------------------------------------------------------------------
__global__ void k_spatial(const float* __restrict__ boxes,
                          float* __restrict__ sf, float* __restrict__ d2)
{
  int idx = blockIdx.x * blockDim.x + threadIdx.x;
  if (idx >= Bz * Nn_ * Nn_) return;
  int j = idx % Nn_, i = (idx / Nn_) % Nn_, b = idx / (Nn_ * Nn_);
  const float* bi = boxes + ((size_t)b * Nn_ + i) * 4;
  const float* bj = boxes + ((size_t)b * Nn_ + j) * 4;
  float cix = (bi[0] + bi[2]) * 0.5f, ciy = (bi[1] + bi[3]) * 0.5f;
  float cjx = (bj[0] + bj[2]) * 0.5f, cjy = (bj[1] + bj[3]) * 0.5f;
  float szix = bi[2] - bi[0], sziy = bi[3] - bi[1];
  float szjx = bj[2] - bj[0], szjy = bj[3] - bj[1];
  float relx = cjx - cix, rely = cjy - ciy;
  float dist = sqrtf(relx * relx + rely * rely);
  float ang  = atan2f(rely, relx);
  float rsx = szjx / (szix + 1e-6f), rsy = szjy / (sziy + 1e-6f);
  float ix = fmaxf(fminf(bi[2], bj[2]) - fmaxf(bi[0], bj[0]), 0.f);
  float iy = fmaxf(fminf(bi[3], bj[3]) - fmaxf(bi[1], bj[1]), 0.f);
  float inter = ix * iy;
  float uni = szix * sziy + szjx * szjy - inter;
  float iou = inter / (uni + 1e-6f);
  float* o = sf + (size_t)idx * SDd;
  o[0] = relx; o[1] = rely; o[2] = dist; o[3] = ang;
  o[4] = rsx;  o[5] = rsy;  o[6] = iou;
  d2[idx] = dist;
}

// K+1 smallest distances per (b,i), stable ties (matches jax top_k on -d2);
// drop the self (slot 0), keep neighbors 1..K.
__global__ void k_topk(const float* __restrict__ d2, int* __restrict__ dsti)
{
  int idx = blockIdx.x * blockDim.x + threadIdx.x;
  if (idx >= Bz * Nn_) return;
  int b = idx / Nn_, i = idx % Nn_;
  const float* row = d2 + (size_t)(b * Nn_ + i) * Nn_;
  unsigned long long used = 0ull;
  for (int s = 0; s <= Kk; ++s) {
    float best = INFINITY; int bj = 0;
    for (int j = 0; j < Nn_; ++j) {
      if (used & (1ull << j)) continue;
      float v = row[j];
      if (v < best) { best = v; bj = j; }
    }
    used |= (1ull << bj);
    if (s >= 1) dsti[(size_t)b * Ee + i * Kk + (s - 1)] = bj;
  }
}

// Edge MLP layer 1: hid = relu(sf[b,src,dst] @ sw1^T + sb1)   (7 -> 64)
__global__ void k_edge_hidden(const float* __restrict__ sf, const int* __restrict__ dsti,
                              const float* __restrict__ sw1, const float* __restrict__ sb1,
                              float* __restrict__ hid)
{
  int idx = blockIdx.x * blockDim.x + threadIdx.x;
  if (idx >= Bz * Ee * 64) return;
  int o = idx % 64, e = (idx / 64) % Ee, b = idx / (64 * Ee);
  int srcn = e / Kk;
  int dn = dsti[(size_t)b * Ee + e];
  const float* ef = sf + (((size_t)b * Nn_ + srcn) * Nn_ + dn) * SDd;
  float s = sb1[o];
  for (int q = 0; q < SDd; ++q) s += ef[q] * sw1[o * SDd + q];
  hid[idx] = fmaxf(s, 0.f);
}

// LayerNorm over D, then relu, optional residual add:  out = res + relu(LN(in))
__global__ void k_ln_relu(const float* __restrict__ in, const float* __restrict__ g,
                          const float* __restrict__ be, const float* __restrict__ res,
                          float* __restrict__ out)
{
  int row = blockIdx.x;
  int tid = threadIdx.x; // 256
  __shared__ float ssum[256], ssq[256];
  const float* r = in + (size_t)row * Dd;
  float s = 0.f, q = 0.f;
  for (int d = tid; d < Dd; d += 256) { float v = r[d]; s += v; q += v * v; }
  ssum[tid] = s; ssq[tid] = q; __syncthreads();
  for (int o = 128; o > 0; o >>= 1) {
    if (tid < o) { ssum[tid] += ssum[tid + o]; ssq[tid] += ssq[tid + o]; }
    __syncthreads();
  }
  float mean = ssum[0] / Dd;
  float var  = ssq[0] / Dd - mean * mean;
  float inv  = rsqrtf(var + 1e-5f);
  for (int d = tid; d < Dd; d += 256) {
    float v = (r[d] - mean) * inv * g[d] + be[d];
    v = v > 0.f ? v : 0.f;
    if (res) v += res[(size_t)row * Dd + d];
    out[(size_t)row * Dd + d] = v;
  }
}

// GAT edge logits: lg[b,e,h] = sum_c leaky(xl[src]+xr[dst]+ew[e],0.2)[h,c]*att[h,c]
__global__ void k_edge_lg(const float* __restrict__ xl, const float* __restrict__ xr,
                          const float* __restrict__ ew, const int* __restrict__ dsti,
                          const float* __restrict__ att, float* __restrict__ lg)
{
  int idx = blockIdx.x * blockDim.x + threadIdx.x;
  if (idx >= Bz * Ee * Hh) return;
  int h = idx % Hh, e = (idx / Hh) % Ee, b = idx / (Hh * Ee);
  int srcn = e / Kk;
  int dn = dsti[(size_t)b * Ee + e];
  const float* xlp = xl + ((size_t)b * Nn_ + srcn) * Dd + h * Cc;
  const float* xrp = xr + ((size_t)b * Nn_ + dn)  * Dd + h * Cc;
  const float* ewp = ew + ((size_t)b * Ee + e)    * Dd + h * Cc;
  const float* ap  = att + h * Cc;
  float s = 0.f;
  for (int c = 0; c < Cc; ++c) {
    float m = xlp[c] + xrp[c] + ewp[c];
    m = m > 0.f ? m : 0.2f * m;
    s += m * ap[c];
  }
  lg[idx] = s;
}

// Segment max + sum(exp) over incoming edges per (b,node,h)
__global__ void k_segstats(const float* __restrict__ lg, const int* __restrict__ dsti,
                           float* __restrict__ mx, float* __restrict__ den)
{
  int idx = blockIdx.x * blockDim.x + threadIdx.x;
  if (idx >= Bz * Nn_ * Hh) return;
  int h = idx % Hh, nnode = (idx / Hh) % Nn_, b = idx / (Hh * Nn_);
  const int*   dr = dsti + (size_t)b * Ee;
  const float* lr = lg   + (size_t)b * Ee * Hh;
  float m = -INFINITY;
  for (int e = 0; e < Ee; ++e) if (dr[e] == nnode) m = fmaxf(m, lr[e * Hh + h]);
  float s = 0.f;
  for (int e = 0; e < Ee; ++e) if (dr[e] == nnode) s += expf(lr[e * Hh + h] - m);
  mx[idx] = m; den[idx] = s;
}

__global__ void k_alpha(const float* __restrict__ lg, const int* __restrict__ dsti,
                        const float* __restrict__ mx, const float* __restrict__ den,
                        float* __restrict__ alpha)
{
  int idx = blockIdx.x * blockDim.x + threadIdx.x;
  if (idx >= Bz * Ee * Hh) return;
  int h = idx % Hh, e = (idx / Hh) % Ee, b = idx / (Hh * Ee);
  int dn = dsti[(size_t)b * Ee + e];
  float m = mx[((size_t)b * Nn_ + dn) * Hh + h];
  float d = den[((size_t)b * Nn_ + dn) * Hh + h];
  alpha[idx] = expf(lg[idx] - m) / (d + 1e-16f);
}

// xn[b,n,ch] = sum_{e: dst==n} alpha[e,h]*xl[src,ch] + bias[ch]
__global__ void k_aggregate(const float* __restrict__ xl, const int* __restrict__ dsti,
                            const float* __restrict__ alpha, const float* __restrict__ bias,
                            float* __restrict__ xn)
{
  int idx = blockIdx.x * blockDim.x + threadIdx.x;
  if (idx >= Bz * Nn_ * Dd) return;
  int ch = idx % Dd, nnode = (idx / Dd) % Nn_, b = idx / (Dd * Nn_);
  int h = ch / Cc;
  const int*   dr = dsti  + (size_t)b * Ee;
  const float* ar = alpha + (size_t)b * Ee * Hh;
  float acc = 0.f;
  for (int e = 0; e < Ee; ++e) {
    if (dr[e] == nnode)
      acc += ar[e * Hh + h] * xl[((size_t)b * Nn_ + (e / Kk)) * Dd + ch];
  }
  xn[idx] = acc + bias[ch];
}

// Mean-pool over nodes; init LSTM state (h0=h1=gf, c0=c1=0)
__global__ void k_pool_init(const float* __restrict__ x, float* __restrict__ h0,
                            float* __restrict__ c0, float* __restrict__ h1,
                            float* __restrict__ c1)
{
  int idx = blockIdx.x * blockDim.x + threadIdx.x;
  if (idx >= Bz * Dd) return;
  int b = idx / Dd, d = idx % Dd;
  float s = 0.f;
  for (int n = 0; n < Nn_; ++n) s += x[((size_t)b * Nn_ + n) * Dd + d];
  s /= (float)Nn_;
  h0[idx] = s; h1[idx] = s; c0[idx] = 0.f; c1[idx] = 0.f;
}

// LSTM pointwise cell from precomputed gates g[b, 4D] (i,f,g,o order)
__global__ void k_lstm_cell(const float* __restrict__ g, float* __restrict__ h,
                            float* __restrict__ c, float* __restrict__ h2, int h2stride)
{
  int idx = blockIdx.x * blockDim.x + threadIdx.x;
  if (idx >= Bz * Dd) return;
  int b = idx / Dd, d = idx % Dd;
  const float* gr = g + (size_t)b * 4 * Dd;
  float ig = gr[d], fg = gr[Dd + d], gg = gr[2 * Dd + d], og = gr[3 * Dd + d];
  float si = 1.f / (1.f + expf(-ig));
  float sf_ = 1.f / (1.f + expf(-fg));
  float so = 1.f / (1.f + expf(-og));
  float cc = sf_ * c[idx] + si * tanhf(gg);
  float hh = so * tanhf(cc);
  c[idx] = cc; h[idx] = hh;
  if (h2) h2[(size_t)b * h2stride + d] = hh;
}

// ---------------------------------------------------------------------------
static void launch_gemm(hipStream_t st, const float* A, int lda,
                        const int* rowIdx, int ridxStride,
                        const _Float16* Wh, const float* bias, float* Y,
                        int M, int Nc, int Kd, int accum)
{
  dim3 grid((Nc / 16 + 3) / 4, M / 32);
  k_wmma_gemm<<<grid, 128, 0, st>>>(A, lda, rowIdx, ridxStride, Wh, Kd, bias,
                                    Y, Nc, M, Nc, Kd, accum);
}

extern "C" void kernel_launch(void* const* d_in, const int* in_sizes, int n_in,
                              void* d_out, int out_size, void* d_ws, size_t ws_size,
                              hipStream_t stream)
{
  (void)in_sizes; (void)n_in; (void)out_size; (void)ws_size;

  const float* nf       = (const float*)d_in[0];
  const float* boxes    = (const float*)d_in[1];
  const int*   caps     = (const int*)  d_in[2];
  const float* sw1      = (const float*)d_in[3];
  const float* sb1      = (const float*)d_in[4];
  const float* sw2      = (const float*)d_in[5];
  const float* sb2      = (const float*)d_in[6];
  const float* ne_w     = (const float*)d_in[7];
  const float* ne_b     = (const float*)d_in[8];
  const float* ne_g     = (const float*)d_in[9];
  const float* ne_beta  = (const float*)d_in[10];
  const float* gat_wl   = (const float*)d_in[11];
  const float* gat_bl   = (const float*)d_in[12];
  const float* gat_wr   = (const float*)d_in[13];
  const float* gat_br   = (const float*)d_in[14];
  const float* gat_we   = (const float*)d_in[15];
  const float* gat_att  = (const float*)d_in[16];
  const float* gat_bias = (const float*)d_in[17];
  const float* ln_g     = (const float*)d_in[18];
  const float* ln_b     = (const float*)d_in[19];
  const float* wih0     = (const float*)d_in[20];
  const float* whh0     = (const float*)d_in[21];
  const float* bih0     = (const float*)d_in[22];
  const float* bhh0     = (const float*)d_in[23];
  const float* wih1     = (const float*)d_in[24];
  const float* whh1     = (const float*)d_in[25];
  const float* bih1     = (const float*)d_in[26];
  const float* bhh1     = (const float*)d_in[27];
  const float* embt     = (const float*)d_in[28];
  const float* fcw      = (const float*)d_in[29];
  const float* fcb      = (const float*)d_in[30];
  float* out = (float*)d_out;

  // --- workspace carve-up (byte-based, 128B aligned for v16h loads) ---
  char* base = (char*)d_ws;
  size_t off = 0;
  auto takeb = [&](size_t bytes) -> void* {
    void* p = base + off; off += (bytes + 127) & ~(size_t)127; return p;
  };
  auto takef = [&](size_t n) -> float* { return (float*)takeb(n * 4); };
  auto takeh = [&](size_t n) -> _Float16* { return (_Float16*)takeb(n * 2); };

  float* sf    = takef((size_t)Bz * Nn_ * Nn_ * SDd);
  float* d2b   = takef((size_t)Bz * Nn_ * Nn_);
  int*   dsti  = (int*)takeb((size_t)Bz * Ee * 4);
  float* hid   = takef((size_t)Bz * Ee * 64);
  float* efeat = takef((size_t)Bz * Ee * Dd);
  float* xbuf  = takef((size_t)Bz * Nn_ * Dd);
  float* tmp   = takef((size_t)Bz * Nn_ * Dd);
  float* xlb   = takef((size_t)Bz * Nn_ * Dd);
  float* xrb   = takef((size_t)Bz * Nn_ * Dd);
  float* ewb   = takef((size_t)Bz * Ee * Dd);
  float* lgb   = takef((size_t)Bz * Ee * Hh);
  float* mxb   = takef((size_t)Bz * Nn_ * Hh);
  float* denb  = takef((size_t)Bz * Nn_ * Hh);
  float* alpb  = takef((size_t)Bz * Ee * Hh);
  float* h0    = takef((size_t)Bz * Dd);
  float* c0    = takef((size_t)Bz * Dd);
  float* h1    = takef((size_t)Bz * Dd);
  float* c1    = takef((size_t)Bz * Dd);
  float* g0    = takef((size_t)Bz * 4 * Dd);
  float* g1    = takef((size_t)Bz * 4 * Dd);
  float* hs    = takef((size_t)Bz * Tt * Dd);
  // f16 weight mirrors (converted once per launch)
  _Float16* sw2h  = takeh((size_t)Dd * 64);
  _Float16* ne_wh = takeh((size_t)Dd * Dd);
  _Float16* wlh   = takeh((size_t)Ll * Dd * Dd);
  _Float16* wrh   = takeh((size_t)Ll * Dd * Dd);
  _Float16* weh   = takeh((size_t)Ll * Dd * Dd);
  _Float16* wih0h = takeh((size_t)4 * Dd * EMBd);
  _Float16* whh0h = takeh((size_t)4 * Dd * Dd);
  _Float16* wih1h = takeh((size_t)4 * Dd * Dd);
  _Float16* whh1h = takeh((size_t)4 * Dd * Dd);
  _Float16* fcwh  = takeh((size_t)VOC * Dd);

  const int TB = 256;
  auto blocks = [](int total) { return (total + 255) / 256; };

  // 0) weight conversion to f16 (halves weight bandwidth; B frag = 1 vec load)
  auto cvt = [&](const float* s, _Float16* d, int n) {
    k_f32_to_f16<<<blocks(n), TB, 0, stream>>>(s, d, n);
  };
  cvt(sw2,  sw2h,  Dd * 64);
  cvt(ne_w, ne_wh, Dd * Dd);
  cvt(gat_wl, wlh, Ll * Dd * Dd);
  cvt(gat_wr, wrh, Ll * Dd * Dd);
  cvt(gat_we, weh, Ll * Dd * Dd);
  cvt(wih0, wih0h, 4 * Dd * EMBd);
  cvt(whh0, whh0h, 4 * Dd * Dd);
  cvt(wih1, wih1h, 4 * Dd * Dd);
  cvt(whh1, whh1h, 4 * Dd * Dd);
  cvt(fcw,  fcwh,  VOC * Dd);

  // 1) spatial features + KNN graph
  k_spatial<<<blocks(Bz * Nn_ * Nn_), TB, 0, stream>>>(boxes, sf, d2b);
  k_topk<<<blocks(Bz * Nn_), TB, 0, stream>>>(d2b, dsti);

  // 2) edge MLP: 7 -> 64 (VALU) -> 512 (WMMA)
  k_edge_hidden<<<blocks(Bz * Ee * 64), TB, 0, stream>>>(sf, dsti, sw1, sb1, hid);
  launch_gemm(stream, hid, 64, nullptr, 0, sw2h, sb2, efeat, Bz * Ee, Dd, 64, 0);

  // 3) node encoder: x = relu(LN(nf @ ne_w^T + ne_b))
  launch_gemm(stream, nf, Dd, nullptr, 0, ne_wh, ne_b, tmp, Bz * Nn_, Dd, Dd, 0);
  k_ln_relu<<<Bz * Nn_, TB, 0, stream>>>(tmp, ne_g, ne_beta, nullptr, xbuf);

  // 4) GAT layers
  for (int l = 0; l < Ll; ++l) {
    launch_gemm(stream, xbuf, Dd, nullptr, 0, wlh + (size_t)l * Dd * Dd,
                gat_bl + (size_t)l * Dd, xlb, Bz * Nn_, Dd, Dd, 0);
    launch_gemm(stream, xbuf, Dd, nullptr, 0, wrh + (size_t)l * Dd * Dd,
                gat_br + (size_t)l * Dd, xrb, Bz * Nn_, Dd, Dd, 0);
    launch_gemm(stream, efeat, Dd, nullptr, 0, weh + (size_t)l * Dd * Dd,
                nullptr, ewb, Bz * Ee, Dd, Dd, 0);
    k_edge_lg<<<blocks(Bz * Ee * Hh), TB, 0, stream>>>(
        xlb, xrb, ewb, dsti, gat_att + (size_t)l * Hh * Cc, lgb);
    k_segstats<<<blocks(Bz * Nn_ * Hh), TB, 0, stream>>>(lgb, dsti, mxb, denb);
    k_alpha<<<blocks(Bz * Ee * Hh), TB, 0, stream>>>(lgb, dsti, mxb, denb, alpb);
    k_aggregate<<<blocks(Bz * Nn_ * Dd), TB, 0, stream>>>(
        xlb, dsti, alpb, gat_bias + (size_t)l * Dd, tmp);
    k_ln_relu<<<Bz * Nn_, TB, 0, stream>>>(tmp, ln_g + (size_t)l * Dd,
                                           ln_b + (size_t)l * Dd, xbuf, xbuf);
  }

  // 5) mean pool + LSTM init
  k_pool_init<<<blocks(Bz * Dd), TB, 0, stream>>>(xbuf, h0, c0, h1, c1);

  // 6) 2-layer LSTM, T=20 sequential steps (gates via WMMA, gather via rowIdx)
  for (int t = 0; t < Tt; ++t) {
    launch_gemm(stream, embt, EMBd, caps + t, Tt, wih0h, bih0, g0, Bz, 4 * Dd, EMBd, 0);
    launch_gemm(stream, h0, Dd, nullptr, 0, whh0h, bhh0, g0, Bz, 4 * Dd, Dd, 1);
    k_lstm_cell<<<blocks(Bz * Dd), TB, 0, stream>>>(g0, h0, c0, nullptr, 0);
    launch_gemm(stream, h0, Dd, nullptr, 0, wih1h, bih1, g1, Bz, 4 * Dd, Dd, 0);
    launch_gemm(stream, h1, Dd, nullptr, 0, whh1h, bhh1, g1, Bz, 4 * Dd, Dd, 1);
    k_lstm_cell<<<blocks(Bz * Dd), TB, 0, stream>>>(g1, h1, c1,
                                                    hs + (size_t)t * Dd, Tt * Dd);
  }

  // 7) vocab projection: (B*T, 512) @ (30000, 512)^T + fcb  -> d_out
  launch_gemm(stream, hs, Dd, nullptr, 0, fcwh, fcb, out, Bz * Tt, VOC, Dd, 0);
}
